// MyModel_38568806318670
// MI455X (gfx1250) — compile-verified, matrix operand order checked by reading, not run on hardware
//
#include <hip/hip_runtime.h>
#include <cstdint>
#include <cstddef>

// ---------------------------------------------------------------------------
// Problem constants (from reference): B=64, L=512, H=1024
// ---------------------------------------------------------------------------
#define SEP_ID 102
#define BB 64
#define LL 512
#define HH 1024

typedef __attribute__((ext_vector_type(16))) _Float16 v16h;
typedef __attribute__((ext_vector_type(8)))  _Float16 v8h;
typedef __attribute__((ext_vector_type(4)))  _Float16 v4h;
typedef __attribute__((ext_vector_type(8)))  float    v8f;

// ---------------------------------------------------------------------------
// Kernel 1: find 2nd and 3rd SEP positions per row (64 rows x 512 ids).
// ---------------------------------------------------------------------------
__global__ __launch_bounds__(64) void find_sep_kernel(const int* __restrict__ ids,
                                                      int* __restrict__ p12) {
    int b = threadIdx.x;
    if (b >= BB) return;
    int cnt = 0, p1 = LL, p2 = LL;
    const int* row = ids + b * LL;
    for (int l = 0; l < LL; ++l) {
        if (row[l] == SEP_ID) {
            ++cnt;
            if (cnt == 2) p1 = l;
            else if (cnt == 3) { p2 = l; break; }
        }
    }
    p12[b * 2]     = p1;
    p12[b * 2 + 1] = p2;
}

// ---------------------------------------------------------------------------
// Kernel 2: masked max-pool of hidden_states -> combined (B, 2H) in f16.
// HBM-bound: 128 MB streamed, coalesced across h. combined is only consumed
// as a WMMA A-operand, so we emit f16 directly.
// ---------------------------------------------------------------------------
__global__ __launch_bounds__(256) void maxpool_kernel(const float* __restrict__ hs,
                                                      const int* __restrict__ p12,
                                                      _Float16* __restrict__ combined_h) {
    int b   = blockIdx.x;
    int col = blockIdx.y * blockDim.x + threadIdx.x;   // 0 .. 2H-1
    int p1 = p12[b * 2], p2 = p12[b * 2 + 1];
    int lo, hi, h;
    if (col < HH) { lo = 1;  hi = p1; h = col; }
    else          { lo = p1; hi = p2; h = col - HH; }
    float m = -INFINITY;
    const float* base = hs + (size_t)b * LL * HH + h;
    for (int l = lo; l < hi; ++l)
        m = fmaxf(m, base[(size_t)l * HH]);
    combined_h[(size_t)b * 2 * HH + col] = (_Float16)m;
}

// ---------------------------------------------------------------------------
// Kernel 3: bulk f32 -> f16 conversion (weights converted exactly once;
// halves the GEMM weight traffic and removes cvt chains from the K-loop).
// float4 in, v4h out -> b128 load / b64 store per thread.
// ---------------------------------------------------------------------------
__global__ __launch_bounds__(256) void f32_to_f16_kernel(const float* __restrict__ src,
                                                         _Float16* __restrict__ dst,
                                                         int n4) {
    int i = blockIdx.x * blockDim.x + threadIdx.x;
    if (i >= n4) return;
    const float4 v = ((const float4*)src)[i];
    v4h o;
    o[0] = (_Float16)v.x; o[1] = (_Float16)v.y;
    o[2] = (_Float16)v.z; o[3] = (_Float16)v.w;
    ((v4h*)dst)[i] = o;
}

// ---------------------------------------------------------------------------
// Kernel 4: WMMA GEMM, f16 operands, f32 accumulation.
//   C[m][n] = sum_k A[m][k] * W[n][k] + bias0[n] (+bias1[n])
// One wave computes a 16x64 tile of C: 4 independent accumulator chains,
// A-fragment reused by 4 WMMAs per K-step. Fragments load as 2x b128.
//
// Fragment layouts per CDNA5 ISA 7.12.2 (wave32):
//   A (16x32 f16): lane<16 -> row M=lane,    K = {0..7, 16..23}
//                  lane>=16 -> row M=lane-16, K = {8..15, 24..31}
//   B (32x16 f16): lane holds row n0+lane%16 of W (= column of B), same split.
//   D (16x16 f32): VGPR r -> M = r + 8*(lane>=16), N = n0 + lane%16.
// ---------------------------------------------------------------------------
__device__ __forceinline__ v16h load_frag(const _Float16* p, int kb) {
    v8h lo = *(const v8h*)(p + kb);
    v8h hi = *(const v8h*)(p + kb + 16);
    return __builtin_shufflevector(lo, hi, 0, 1, 2, 3, 4, 5, 6, 7,
                                           8, 9, 10, 11, 12, 13, 14, 15);
}

__global__ __launch_bounds__(32) void wmma_gemm_kernel(const _Float16* __restrict__ A,
                                                       const _Float16* __restrict__ W,
                                                       const float* __restrict__ bias0,
                                                       const float* __restrict__ bias1,
                                                       float* __restrict__ C,
                                                       int Ncols, int K) {
    const int m0   = blockIdx.x * 16;
    const int n0   = blockIdx.y * 64;
    const int lane = threadIdx.x;        // 0..31
    const int half = lane >> 4;          // K-split selector
    const int l16  = lane & 15;

    v8f acc0 = {}, acc1 = {}, acc2 = {}, acc3 = {};
    const _Float16* arow = A + (size_t)(m0 + l16) * K;
    const _Float16* wr0  = W + (size_t)(n0 + l16) * K;
    const _Float16* wr1  = wr0 + (size_t)16 * K;
    const _Float16* wr2  = wr0 + (size_t)32 * K;
    const _Float16* wr3  = wr0 + (size_t)48 * K;

#pragma unroll 2
    for (int k0 = 0; k0 < K; k0 += 32) {
        const int kb = k0 + half * 8;
        const v16h af = load_frag(arow, kb);
        const v16h b0 = load_frag(wr0, kb);
        const v16h b1 = load_frag(wr1, kb);
        const v16h b2 = load_frag(wr2, kb);
        const v16h b3 = load_frag(wr3, kb);
        acc0 = __builtin_amdgcn_wmma_f32_16x16x32_f16(false, af, false, b0,
                                                      (short)0, acc0, false, false);
        acc1 = __builtin_amdgcn_wmma_f32_16x16x32_f16(false, af, false, b1,
                                                      (short)0, acc1, false, false);
        acc2 = __builtin_amdgcn_wmma_f32_16x16x32_f16(false, af, false, b2,
                                                      (short)0, acc2, false, false);
        acc3 = __builtin_amdgcn_wmma_f32_16x16x32_f16(false, af, false, b3,
                                                      (short)0, acc3, false, false);
    }

#pragma unroll
    for (int tile = 0; tile < 4; ++tile) {
        const v8f acc = tile == 0 ? acc0 : tile == 1 ? acc1 : tile == 2 ? acc2 : acc3;
        const int n = n0 + tile * 16 + l16;
        float bb = bias0[n];
        if (bias1) bb += bias1[n];
#pragma unroll
        for (int r = 0; r < 8; ++r) {
            const int m = m0 + r + half * 8;
            C[(size_t)m * Ncols + n] = acc[r] + bb;
        }
    }
}

// ---------------------------------------------------------------------------
// Kernel 5: sequential LSTM recurrence (64 steps over the batch axis).
// Gates G = combined @ w_ih.T + b_ih + b_hh precomputed by WMMA GEMM, so the
// sequential critical path is only h @ w_hh.T per step. One 1024-thread
// workgroup per direction; h in LDS (float4 broadcast reads = ds_load_b128,
// conflict-free), c in registers. Output written directly as f16 since it is
// only consumed as the next GEMM's A-operand.
// ---------------------------------------------------------------------------
__global__ __launch_bounds__(1024) void lstm_kernel(const float* __restrict__ G,
                                                    const float* __restrict__ w_hh,
                                                    _Float16* __restrict__ lstm_out_h,
                                                    int reverse) {
    __shared__ float h_sh[HH];
    const int j = threadIdx.x;           // owned hidden unit
    float c = 0.0f;
    h_sh[j] = 0.0f;
    __syncthreads();

    const float4* wi4 = (const float4*)(w_hh + (size_t)(j)          * HH);
    const float4* wf4 = (const float4*)(w_hh + (size_t)(j + HH)     * HH);
    const float4* wg4 = (const float4*)(w_hh + (size_t)(j + 2 * HH) * HH);
    const float4* wo4 = (const float4*)(w_hh + (size_t)(j + 3 * HH) * HH);
    const float4* h4  = (const float4*)h_sh;

    for (int t = 0; t < BB; ++t) {
        const int tt = reverse ? (BB - 1 - t) : t;
        const float* g0 = G + (size_t)tt * 4 * HH;

        // Pull next step's gate pre-activations toward the caches early.
        const int tn = reverse ? (BB - 2 - t) : (t + 1);
        if (tn >= 0 && tn < BB)
            __builtin_prefetch(G + (size_t)tn * 4 * HH + j, 0, 0);

        float gi = g0[j], gf = g0[j + HH], gg = g0[j + 2 * HH], go = g0[j + 3 * HH];
#pragma unroll 2
        for (int k4 = 0; k4 < HH / 4; ++k4) {
            const float4 hv = h4[k4];
            const float4 a = wi4[k4];
            gi = fmaf(a.x, hv.x, fmaf(a.y, hv.y, fmaf(a.z, hv.z, fmaf(a.w, hv.w, gi))));
            const float4 b = wf4[k4];
            gf = fmaf(b.x, hv.x, fmaf(b.y, hv.y, fmaf(b.z, hv.z, fmaf(b.w, hv.w, gf))));
            const float4 g = wg4[k4];
            gg = fmaf(g.x, hv.x, fmaf(g.y, hv.y, fmaf(g.z, hv.z, fmaf(g.w, hv.w, gg))));
            const float4 o = wo4[k4];
            go = fmaf(o.x, hv.x, fmaf(o.y, hv.y, fmaf(o.z, hv.z, fmaf(o.w, hv.w, go))));
        }
        const float si = 1.0f / (1.0f + __expf(-gi));
        const float sf = 1.0f / (1.0f + __expf(-gf));
        const float so = 1.0f / (1.0f + __expf(-go));
        c = sf * c + si * tanhf(gg);
        const float hnew = so * tanhf(c);

        __syncthreads();                 // all dots done before h update
        h_sh[j] = hnew;
        lstm_out_h[(size_t)tt * 2 * HH + reverse * HH + j] = (_Float16)hnew;
        __syncthreads();                 // h updated before next step's dots
    }
}

// ---------------------------------------------------------------------------
// Kernel 6: final tiny projection  logits = out1 @ w2.T + b2  (64 x 2).
// ---------------------------------------------------------------------------
__global__ __launch_bounds__(256) void logits_kernel(const float* __restrict__ out1,
                                                     const float* __restrict__ w2,
                                                     const float* __restrict__ b2,
                                                     float* __restrict__ logits) {
    __shared__ float red0[256];
    __shared__ float red1[256];
    const int b = blockIdx.x, t = threadIdx.x;
    const float* row = out1 + (size_t)b * 2 * HH;
    float s0 = 0.0f, s1 = 0.0f;
    for (int k = t; k < 2 * HH; k += 256) {
        const float v = row[k];
        s0 = fmaf(v, w2[k], s0);
        s1 = fmaf(v, w2[2 * HH + k], s1);
    }
    red0[t] = s0; red1[t] = s1;
    __syncthreads();
    for (int off = 128; off > 0; off >>= 1) {
        if (t < off) { red0[t] += red0[t + off]; red1[t] += red1[t + off]; }
        __syncthreads();
    }
    if (t == 0) {
        logits[b * 2]     = red0[0] + b2[0];
        logits[b * 2 + 1] = red1[0] + b2[1];
    }
}

// ---------------------------------------------------------------------------
// Host-side orchestration.
// Workspace layout (offsets in bytes):
//   combined_h : 64*2048*2   = 256 KB @ 0x0000000
//   G_f        : 64*4096*4   =   1 MB @ 0x0040000
//   G_b        : 64*4096*4   =   1 MB @ 0x0140000
//   lstm_out_h : 64*2048*2   = 256 KB @ 0x0240000
//   out1       : 64*2048*4   = 512 KB @ 0x0280000
//   p12        : 64*2*4      = 512  B @ 0x0300000
//   wihf_h     : 4096*2048*2 =  16 MB @ 0x0400000
//   wihb_h     : 4096*2048*2 =  16 MB @ 0x1400000
//   w1_h       : 2048*2048*2 =   8 MB @ 0x2400000
// ---------------------------------------------------------------------------
extern "C" void kernel_launch(void* const* d_in, const int* in_sizes, int n_in,
                              void* d_out, int out_size, void* d_ws, size_t ws_size,
                              hipStream_t stream) {
    const int*   input_ids = (const int*)  d_in[0];
    const float* hidden    = (const float*)d_in[1];
    const float* w_ih_f    = (const float*)d_in[2];
    const float* w_hh_f    = (const float*)d_in[3];
    const float* b_ih_f    = (const float*)d_in[4];
    const float* b_hh_f    = (const float*)d_in[5];
    const float* w_ih_b    = (const float*)d_in[6];
    const float* w_hh_b    = (const float*)d_in[7];
    const float* b_ih_b    = (const float*)d_in[8];
    const float* b_hh_b    = (const float*)d_in[9];
    const float* w1        = (const float*)d_in[10];
    const float* b1        = (const float*)d_in[11];
    const float* w2        = (const float*)d_in[12];
    const float* b2        = (const float*)d_in[13];
    float* logits = (float*)d_out;

    char* ws = (char*)d_ws;
    _Float16* combined_h = (_Float16*)(ws + 0x0000000);
    float*    G_f        = (float*)   (ws + 0x0040000);
    float*    G_b        = (float*)   (ws + 0x0140000);
    _Float16* lstm_out_h = (_Float16*)(ws + 0x0240000);
    float*    out1       = (float*)   (ws + 0x0280000);
    int*      p12        = (int*)     (ws + 0x0300000);
    _Float16* wihf_h     = (_Float16*)(ws + 0x0400000);
    _Float16* wihb_h     = (_Float16*)(ws + 0x1400000);
    _Float16* w1_h       = (_Float16*)(ws + 0x2400000);

    // 0) one-time weight conversions f32 -> f16 (streamed, HBM-bound)
    {
        const int n4_ih = (4 * HH * 2 * HH) / 4;   // 2,097,152 float4s
        f32_to_f16_kernel<<<(n4_ih + 255) / 256, 256, 0, stream>>>(w_ih_f, wihf_h, n4_ih);
        f32_to_f16_kernel<<<(n4_ih + 255) / 256, 256, 0, stream>>>(w_ih_b, wihb_h, n4_ih);
        const int n4_w1 = (2 * HH * 2 * HH) / 4;
        f32_to_f16_kernel<<<(n4_w1 + 255) / 256, 256, 0, stream>>>(w1, w1_h, n4_w1);
    }

    // 1) SEP positions
    find_sep_kernel<<<1, 64, 0, stream>>>(input_ids, p12);

    // 2) masked max-pool -> combined_h (64 x 2048, f16)
    maxpool_kernel<<<dim3(BB, (2 * HH) / 256), 256, 0, stream>>>(hidden, p12, combined_h);

    // 3) input projections (hoisted out of the recurrence), WMMA GEMMs
    //    G_dir = combined @ w_ih_dir.T + b_ih_dir + b_hh_dir   (64 x 4096)
    wmma_gemm_kernel<<<dim3(BB / 16, (4 * HH) / 64), 32, 0, stream>>>(
        combined_h, wihf_h, b_ih_f, b_hh_f, G_f, 4 * HH, 2 * HH);
    wmma_gemm_kernel<<<dim3(BB / 16, (4 * HH) / 64), 32, 0, stream>>>(
        combined_h, wihb_h, b_ih_b, b_hh_b, G_b, 4 * HH, 2 * HH);

    // 4) sequential recurrences (forward then backward; independent)
    lstm_kernel<<<1, 1024, 0, stream>>>(G_f, w_hh_f, lstm_out_h, 0);
    lstm_kernel<<<1, 1024, 0, stream>>>(G_b, w_hh_b, lstm_out_h, 1);

    // 5) out1 = lstm_out @ w1.T + b1   (64 x 2048), WMMA GEMM
    wmma_gemm_kernel<<<dim3(BB / 16, (2 * HH) / 64), 32, 0, stream>>>(
        lstm_out_h, w1_h, b1, nullptr, out1, 2 * HH, 2 * HH);

    // 6) logits = out1 @ w2.T + b2    (64 x 2)
    logits_kernel<<<BB, 256, 0, stream>>>(out1, w2, b2, logits);
}
